// MemoryEfficientBSpline_34428457845125
// MI455X (gfx1250) — matrix operand order repaired and checked by reading, not compiled, not run on hardware
//
#include <hip/hip_runtime.h>

typedef __attribute__((ext_vector_type(16))) _Float16 v16h;
typedef __attribute__((ext_vector_type(8)))  _Float16 v8h;
typedef __attribute__((ext_vector_type(2)))  __fp16   v2hp;  // cvt_pkrtz result
typedef __attribute__((ext_vector_type(8)))  float    v8f;

#define B_      8
#define IN_DIM  64
#define OUT_DIM 64
#define HH      192
#define WW      192
#define P_      (HH * WW)          /* 36864 */
#define G_      6
#define KDIM    (IN_DIM * G_)      /* 384 */
#define KSTEPS  (KDIM / 32)        /* 12 */
#define MTILE   128                /* pixels per block */
#define NWAVES  4
#define THREADS (NWAVES * 32)      /* 128 */
#define A_STRIDE (KDIM + 8)        /* 392 halfs per row -> rows 4 banks apart */
#define NTILES_P (P_ / MTILE)      /* 288 */
#define CSTRIDE (MTILE + 4)        /* 132: f32 stride for output staging */

// ---------------------------------------------------------------------------
// Kernel 1: convert coef f32 -> f16 into workspace, layout [b][o][k] (k contig)
// ---------------------------------------------------------------------------
__global__ void coef_to_f16_kernel(const float* __restrict__ coef,
                                   _Float16* __restrict__ coefh) {
    int n = blockIdx.x * blockDim.x + threadIdx.x;   // exact grid, no tail
    coefh[n] = (_Float16)coef[n];
}

// ---------------------------------------------------------------------------
// Kernel 2: hat-basis A-tile + B-slab in LDS, v_wmma_f32_16x16x32_f16 GEMM
// Each wave: 32 pixels x 64 outputs (2 M-subtiles x 4 N-tiles = 8 accums)
// ---------------------------------------------------------------------------
__global__ void __launch_bounds__(THREADS)
bspline_wmma_kernel(const float* __restrict__ x,
                    const _Float16* __restrict__ coefh,
                    float* __restrict__ out) {
    __shared__ __align__(16) _Float16 Ash[MTILE  * A_STRIDE];  // 100352 B
    __shared__ __align__(16) _Float16 Bsh[OUT_DIM * A_STRIDE]; //  50176 B

    const int tile = blockIdx.x;
    const int b    = tile / NTILES_P;
    const int p0   = (tile % NTILES_P) * MTILE;
    const int tid  = threadIdx.x;

    // ---- Phase 0: stage f16 coef slab [o][k] into LDS (once per block) ----
#pragma unroll
    for (int it = 0; it < (OUT_DIM * KDIM / 8) / THREADS; ++it) {
        int c  = it * THREADS + tid;
        int o  = c / (KDIM / 8);
        int kc = (c % (KDIM / 8)) * 8;
        v8h t = *(const v8h*)(coefh + (size_t)(b * OUT_DIM + o) * KDIM + kc);
        *(v8h*)(&Bsh[o * A_STRIDE + kc]) = t;
    }

    // ---- Phase 1: hat-basis weights -> dense f16 A tile in LDS ----
    // Two-hot interp == hat functions: Wt[g] = max(0, 1 - |nx - g|), g=0..5
    // (identical to reference incl. idx clamp / nx==5 edge case)
#pragma unroll
    for (int it = 0; it < (MTILE * IN_DIM / 4) / THREADS; ++it) {
        int slot = it * THREADS + tid;
        int i    = slot >> 5;           // channel (32 float4 per 128-pix row)
        int p4   = slot & 31;           // float4 index within pixel tile

        float4 xv4 = *(const float4*)(x + (size_t)(b * IN_DIM + i) * P_
                                        + p0 + p4 * 4);
        float xs[4] = {xv4.x, xv4.y, xv4.z, xv4.w};
#pragma unroll
        for (int j = 0; j < 4; ++j) {
            float xc = fminf(fmaxf(xs[j], -1.0f), 1.0f);   // v_med3_f32
            float nx = __builtin_fmaf(xc, 2.5f, 2.5f);     // (xc+1)/0.4

            float w[G_];
#pragma unroll
            for (int g = 0; g < G_; ++g) {
                float t = nx - (float)g;                   // v_sub_f32
                w[g] = fmaxf(1.0f - __builtin_fabsf(t), 0.0f); // sub|abs|+max
            }
            // v_cvt_pk_rtz_f16_f32: two weights -> one packed-f16 dword
            v2hp d0 = __builtin_amdgcn_cvt_pkrtz(w[0], w[1]);
            v2hp d1 = __builtin_amdgcn_cvt_pkrtz(w[2], w[3]);
            v2hp d2 = __builtin_amdgcn_cvt_pkrtz(w[4], w[5]);

            int pl = p4 * 4 + j;
            v2hp* dst = (v2hp*)&Ash[pl * A_STRIDE + i * G_];
            dst[0] = d0;
            dst[1] = d1;
            dst[2] = d2;
        }
    }
    __syncthreads();

    // ---- Phase 2: per-wave 32x64 GEMM: 8 WMMAs per 12 ds_load_b128 ----
    const int wave  = tid >> 5;
    const int lane  = tid & 31;
    const int lhalf = lane >> 4;     // 0: k 0-7,16-23 of each step; 1: k 8-15,24-31
    const int nlane = lane & 15;

    // A rows for the wave's two 16-pixel subtiles
    const _Float16* abase0 =
        &Ash[(wave * 32 + 0  + nlane) * A_STRIDE + lhalf * 8];
    const _Float16* abase1 =
        &Ash[(wave * 32 + 16 + nlane) * A_STRIDE + lhalf * 8];
    const _Float16* bbase = &Bsh[lhalf * 8];

    union AV { v16h v; v8h h[2]; };
    auto loadA = [](const _Float16* base, int ks) {
        AV a;
        a.h[0] = *(const v8h*)(base + ks);         // ds_load_b128
        a.h[1] = *(const v8h*)(base + ks + 16);    // ds_load_b128
        return a.v;
    };

    v8f acc[8];
#pragma unroll
    for (int q = 0; q < 8; ++q) acc[q] = (v8f){};

#pragma unroll
    for (int s = 0; s < KSTEPS; ++s) {
        const int ks = s * 32;
        v16h a0 = loadA(abase0, ks);
        v16h a1 = loadA(abase1, ks);
#pragma unroll
        for (int nt = 0; nt < 4; ++nt) {
            AV bf;
            const _Float16* bp = bbase + (nt * 16 + nlane) * A_STRIDE + ks;
            bf.h[0] = *(const v8h*)(bp);           // ds_load_b128
            bf.h[1] = *(const v8h*)(bp + 16);      // ds_load_b128
            // Two WMMAs share one B fragment -> back-to-back issue, one wait
            acc[nt * 2 + 0] = __builtin_amdgcn_wmma_f32_16x16x32_f16(
                false, a0, false, bf.v, (short)0, acc[nt * 2 + 0], false, false);
            acc[nt * 2 + 1] = __builtin_amdgcn_wmma_f32_16x16x32_f16(
                false, a1, false, bf.v, (short)0, acc[nt * 2 + 1], false, false);
        }
    }

    // ---- Phase 3: transpose through LDS for coalesced global stores ----
    __syncthreads();                       // all waves done reading A tile
    float* csh = (float*)Ash;              // reuse A region: [64][CSTRIDE]

#pragma unroll
    for (int r = 0; r < 8; ++r) {
#pragma unroll
        for (int nt = 0; nt < 4; ++nt) {
#pragma unroll
            for (int sub = 0; sub < 2; ++sub) {
                int m = wave * 32 + sub * 16 + lhalf * 8 + r;  // pixel in tile
                csh[(nt * 16 + nlane) * CSTRIDE + m] = acc[nt * 2 + sub][r];
            }
        }
    }
    __syncthreads();

    // One full 128-pixel output row per iteration: perfectly coalesced
#pragma unroll
    for (int o = 0; o < OUT_DIM; ++o) {
        out[(size_t)(b * OUT_DIM + o) * P_ + p0 + tid] = csh[o * CSTRIDE + tid];
    }
}

// ---------------------------------------------------------------------------
extern "C" void kernel_launch(void* const* d_in, const int* in_sizes, int n_in,
                              void* d_out, int out_size, void* d_ws, size_t ws_size,
                              hipStream_t stream) {
    const float* x    = (const float*)d_in[0];   // [8,64,192,192] f32
    const float* coef = (const float*)d_in[1];   // [8,64,64,6]    f32
    float* out        = (float*)d_out;           // [8,64,192,192] f32
    _Float16* coefh   = (_Float16*)d_ws;         // 8*64*384*2 = 384 KB staging

    const int ncoef = B_ * OUT_DIM * KDIM;       // 196608, divisible by 256
    coef_to_f16_kernel<<<ncoef / 256, 256, 0, stream>>>(coef, coefh);

    bspline_wmma_kernel<<<B_ * NTILES_P, THREADS, 0, stream>>>(x, coefh, out);
}
